// GNN_37177236914658
// MI455X (gfx1250) — matrix operand order repaired and checked by reading, not compiled, not run on hardware
//
#include <hip/hip_runtime.h>

#define N_NODES 50000
#define N_EDGES 800000
#define K_DIM   128          // input feature dim for both layers

typedef __attribute__((ext_vector_type(16))) __bf16 v16bf;
typedef __attribute__((ext_vector_type(8)))  float  v8f;
typedef unsigned short u16;
typedef unsigned int   u32;

union V16U { v16bf v; uint4 q[2]; };

// ---- fp32 <-> bf16 helpers (round-to-nearest-even) --------------------------
static __device__ __forceinline__ u16 f2bf(float f) {
    u32 u = __float_as_uint(f);
    u32 r = u + 0x7FFFu + ((u >> 16) & 1u);
    return (u16)(r >> 16);
}
static __device__ __forceinline__ float bf2f(u16 s) {
    return __uint_as_float(((u32)s) << 16);
}

// ---- workspace zeroing (atomics accumulate; must re-zero every launch) ------
__global__ __launch_bounds__(256) void zero_kernel(float* p, long n) {
    long i = (long)blockIdx.x * blockDim.x + threadIdx.x;
    long stride = (long)gridDim.x * blockDim.x;
    for (; i < n; i += stride) p[i] = 0.0f;
}

// ---- edge scatter: one wave32 per edge --------------------------------------
// Coalesced 512B row gather + 512B of contiguous f32 atomic adds (L2-resident).
__global__ __launch_bounds__(256) void scatter_mean_accum(
    const float* __restrict__ X,     // [N_NODES, 128]
    const int*   __restrict__ src,   // [E]
    const int*   __restrict__ dst,   // [E]
    float*       __restrict__ Agg,   // [N_NODES, 128]
    float*       __restrict__ deg,   // [N_NODES] or nullptr
    int E)
{
    int wave = (blockIdx.x * blockDim.x + threadIdx.x) >> 5;
    int lane = threadIdx.x & 31;
    if (wave >= E) return;
    int s = src[wave];
    int d = dst[wave];
    const float4 xv = *(const float4*)(X + (size_t)s * K_DIM + lane * 4);
    float* arow = Agg + (size_t)d * K_DIM + lane * 4;
    atomicAdd(arow + 0, xv.x);
    atomicAdd(arow + 1, xv.y);
    atomicAdd(arow + 2, xv.z);
    atomicAdd(arow + 3, xv.w);
    if (deg != nullptr && lane == 0) atomicAdd(deg + d, 1.0f);
}

// ---- one-pass fp32 -> bf16 hi/lo split (TF32x3-style), optional mean-norm ---
// Each thread converts 4 consecutive elements (row length 128 -> same row).
__global__ __launch_bounds__(256) void split_bf16(
    const float* __restrict__ in,    // [n4*4] fp32, row length 128
    const float* __restrict__ deg,   // per-row degree or nullptr
    u16*         __restrict__ hi,    // [n4*4] bf16 hi
    u16*         __restrict__ lo,    // [n4*4] bf16 residual
    long n4)                         // number of float4 quads
{
    long i = (long)blockIdx.x * blockDim.x + threadIdx.x;
    long stride = (long)gridDim.x * blockDim.x;
    for (; i < n4; i += stride) {
        float4 v = ((const float4*)in)[i];
        float s = 1.0f;
        if (deg != nullptr) s = 1.0f / fmaxf(deg[(i * 4) >> 7], 1.0f);
        float f[4] = { v.x * s, v.y * s, v.z * s, v.w * s };
        u32 hp[2], lp[2];
#pragma unroll
        for (int j = 0; j < 2; ++j) {
            u16 h0 = f2bf(f[2 * j]),     h1 = f2bf(f[2 * j + 1]);
            u16 l0 = f2bf(f[2 * j]     - bf2f(h0));
            u16 l1 = f2bf(f[2 * j + 1] - bf2f(h1));
            hp[j] = (u32)h0 | ((u32)h1 << 16);
            lp[j] = (u32)l0 | ((u32)l1 << 16);
        }
        ((uint2*)hi)[i] = make_uint2(hp[0], hp[1]);
        ((uint2*)lo)[i] = make_uint2(lp[0], lp[1]);
    }
}

// ---- fused SAGE layer GEMM: (aggN@Wl^T + x@Wr^T + b) then ReLU --------------
// All operands pre-split to bf16 hi/lo in row-major [rows,128]; loads map
// directly onto WMMA operand registers (two uint4 loads each, zero VALU prep).
template <int NOUT>
__global__ __launch_bounds__(256) void sage_layer_wmma(
    const u16* __restrict__ xHi, const u16* __restrict__ xLo,   // [N,128]
    const u16* __restrict__ aHi, const u16* __restrict__ aLo,   // [N,128] (normalized agg)
    const u16* __restrict__ wlHi, const u16* __restrict__ wlLo, // [NOUT,128]
    const u16* __restrict__ wrHi, const u16* __restrict__ wrLo, // [NOUT,128]
    const float* __restrict__ bias,                             // [NOUT]
    float*       __restrict__ Out)                              // [N,NOUT]
{
    constexpr int NTN = NOUT / 16;
    const int wave = threadIdx.x >> 5;
    const int lane = threadIdx.x & 31;
    const int tile = blockIdx.x * 8 + wave;           // n-major tiling
    if (tile >= (N_NODES / 16) * NTN) return;         // wave-uniform, never ragged
    const int tile_m = tile / NTN;
    const int tile_n = tile % NTN;

    const int mrow = tile_m * 16 + (lane & 15);       // node row (A operand)
    const int wrow = tile_n * 16 + (lane & 15);       // output feature (B operand)
    const int g    = lane >> 4;

    const size_t aoff = (size_t)mrow * K_DIM;
    const size_t woff = (size_t)wrow * K_DIM;

    v8f acc = {};
#pragma unroll
    for (int kb = 0; kb < K_DIM; kb += 32) {
        // A layout: halves 0..7 -> K=kb+8g+j ; halves 8..15 -> K=kb+16+8g+j
        const size_t a0 = aoff + kb + 8 * g;
        const size_t a1 = aoff + kb + 16 + 8 * g;
        // B layout: same as A: halves 0..7 -> K=kb+8g+j ; halves 8..15 -> K=kb+16+8g+j
        const size_t b0 = woff + kb + 8 * g;

        V16U Ah, Al, Xh, Xl, Lh, Ll, Rh, Rl;
        Ah.q[0] = *(const uint4*)(aHi + a0);  Ah.q[1] = *(const uint4*)(aHi + a1);
        Al.q[0] = *(const uint4*)(aLo + a0);  Al.q[1] = *(const uint4*)(aLo + a1);
        Xh.q[0] = *(const uint4*)(xHi + a0);  Xh.q[1] = *(const uint4*)(xHi + a1);
        Xl.q[0] = *(const uint4*)(xLo + a0);  Xl.q[1] = *(const uint4*)(xLo + a1);
        Lh.q[0] = *(const uint4*)(wlHi + b0); Lh.q[1] = *(const uint4*)(wlHi + b0 + 16);
        Ll.q[0] = *(const uint4*)(wlLo + b0); Ll.q[1] = *(const uint4*)(wlLo + b0 + 16);
        Rh.q[0] = *(const uint4*)(wrHi + b0); Rh.q[1] = *(const uint4*)(wrHi + b0 + 16);
        Rl.q[0] = *(const uint4*)(wrLo + b0); Rl.q[1] = *(const uint4*)(wrLo + b0 + 16);

        // aggN @ Wl^T : hi*hi + hi*lo + lo*hi (fp32 accumulate)
        acc = __builtin_amdgcn_wmma_f32_16x16x32_bf16(false, Ah.v, false, Lh.v, (short)0, acc, false, false);
        acc = __builtin_amdgcn_wmma_f32_16x16x32_bf16(false, Ah.v, false, Ll.v, (short)0, acc, false, false);
        acc = __builtin_amdgcn_wmma_f32_16x16x32_bf16(false, Al.v, false, Lh.v, (short)0, acc, false, false);
        // x @ Wr^T
        acc = __builtin_amdgcn_wmma_f32_16x16x32_bf16(false, Xh.v, false, Rh.v, (short)0, acc, false, false);
        acc = __builtin_amdgcn_wmma_f32_16x16x32_bf16(false, Xh.v, false, Rl.v, (short)0, acc, false, false);
        acc = __builtin_amdgcn_wmma_f32_16x16x32_bf16(false, Xl.v, false, Rh.v, (short)0, acc, false, false);
    }

    // C/D layout: VGPR r -> row tile_m*16 + r + 8g, col = wrow
    const float bn = bias[wrow];
    const int baseRow = tile_m * 16 + 8 * g;
#pragma unroll
    for (int r = 0; r < 8; ++r) {
        float v = acc[r] + bn;
        Out[(size_t)(baseRow + r) * NOUT + wrow] = fmaxf(v, 0.0f);
    }
}

extern "C" void kernel_launch(void* const* d_in, const int* in_sizes, int n_in,
                              void* d_out, int out_size, void* d_ws, size_t ws_size,
                              hipStream_t stream) {
    (void)in_sizes; (void)n_in; (void)out_size; (void)ws_size;
    const float* x   = (const float*)d_in[0];
    const int*   ei  = (const int*)d_in[1];   // [2, E] row-major int32
    const float* W1l = (const float*)d_in[2];
    const float* b1  = (const float*)d_in[3];
    const float* W1r = (const float*)d_in[4];
    const float* W2l = (const float*)d_in[5];
    const float* b2  = (const float*)d_in[6];
    const float* W2r = (const float*)d_in[7];
    float* out = (float*)d_out;

    // ---- workspace layout (bytes) ----
    const size_t NF = (size_t)N_NODES * K_DIM;   // elements in one [N,128] array
    char* p = (char*)d_ws;
    float* agg  = (float*)p;            p += NF * sizeof(float);        // fp32 scatter target
    float* h    = (float*)p;            p += NF * sizeof(float);        // layer-1 output (fp32)
    float* deg  = (float*)p;            p += N_NODES * sizeof(float);
    u16* aHi    = (u16*)p;              p += NF * sizeof(u16);
    u16* aLo    = (u16*)p;              p += NF * sizeof(u16);
    u16* xHi    = (u16*)p;              p += NF * sizeof(u16);
    u16* xLo    = (u16*)p;              p += NF * sizeof(u16);
    u16* wlHi   = (u16*)p;              p += (size_t)256 * K_DIM * sizeof(u16);
    u16* wlLo   = (u16*)p;              p += (size_t)256 * K_DIM * sizeof(u16);
    u16* wrHi   = (u16*)p;              p += (size_t)256 * K_DIM * sizeof(u16);
    u16* wrLo   = (u16*)p;

    const int* src = ei;
    const int* dst = ei + N_EDGES;

    const long n4_feat = (long)(NF / 4);          // quads in a [N,128] array
    const long n4_w1   = (long)(128 * K_DIM / 4);
    const long n4_w2   = (long)(256 * K_DIM / 4);

    // ================= Layer 1 (128 -> 128) =================
    zero_kernel<<<2048, 256, 0, stream>>>(agg, (long)NF + N_NODES); // agg (+deg right after h? no: zero agg and deg separately)
    zero_kernel<<<64,   256, 0, stream>>>(deg, (long)N_NODES);
    scatter_mean_accum<<<(N_EDGES + 7) / 8, 256, 0, stream>>>(x, src, dst, agg, deg, N_EDGES);

    split_bf16<<<2048, 256, 0, stream>>>(agg, deg,    aHi, aLo, n4_feat); // fused mean-normalize
    split_bf16<<<2048, 256, 0, stream>>>(x,   nullptr, xHi, xLo, n4_feat);
    split_bf16<<<64,   256, 0, stream>>>(W1l, nullptr, wlHi, wlLo, n4_w1);
    split_bf16<<<64,   256, 0, stream>>>(W1r, nullptr, wrHi, wrLo, n4_w1);

    sage_layer_wmma<128><<<(N_NODES / 16) * (128 / 16) / 8, 256, 0, stream>>>(
        xHi, xLo, aHi, aLo, wlHi, wlLo, wrHi, wrLo, b1, h);

    // ================= Layer 2 (128 -> 256) =================
    zero_kernel<<<2048, 256, 0, stream>>>(agg, (long)NF);           // deg is reused as-is
    scatter_mean_accum<<<(N_EDGES + 7) / 8, 256, 0, stream>>>(h, src, dst, agg, nullptr, N_EDGES);

    split_bf16<<<2048, 256, 0, stream>>>(agg, deg,    aHi, aLo, n4_feat);
    split_bf16<<<2048, 256, 0, stream>>>(h,   nullptr, xHi, xLo, n4_feat);
    split_bf16<<<64,   256, 0, stream>>>(W2l, nullptr, wlHi, wlLo, n4_w2);
    split_bf16<<<64,   256, 0, stream>>>(W2r, nullptr, wrHi, wrLo, n4_w2);

    sage_layer_wmma<256><<<(N_NODES / 16) * (256 / 16) / 8, 256, 0, stream>>>(
        xHi, xLo, aHi, aLo, wlHi, wlLo, wrHi, wrLo, b2, out);
}
